// WaveRNN_82463372083317
// MI455X (gfx1250) — matrix operand order repaired and checked by reading, not compiled, not run on hardware
//
#include <hip/hip_runtime.h>

// ---------- types ----------
typedef __bf16  bf16x8  __attribute__((ext_vector_type(8)));
typedef __bf16  bf16x16 __attribute__((ext_vector_type(16)));
typedef float   f32x8   __attribute__((ext_vector_type(8)));

#define NB 32768     // batch
#define NH 896       // hidden
#define NSP 448      // split
#define NQ 256       // quantization

static __device__ __forceinline__ __bf16 to_bf16(float f) {
  unsigned u = __float_as_uint(f);
  u += 0x7fffu + ((u >> 16) & 1u);     // round-to-nearest-even
  unsigned short s = (unsigned short)(u >> 16);
  __bf16 b;
  __builtin_memcpy(&b, &s, 2);
  return b;
}

static __device__ __forceinline__ f32x8 wmma_bf16(bf16x16 a, bf16x16 b, f32x8 c) {
  return __builtin_amdgcn_wmma_f32_16x16x32_bf16(false, a, false, b, (short)0, c, false, false);
}

// CDNA5 async global->LDS (ASYNCcnt-tracked, bypasses VGPRs).
// VDST = wave-relative LDS byte offset (= low 32 bits of generic shared ptr),
// VADDR = 64-bit global address, saddr=off.
static __device__ __forceinline__ void async_ld_b128(void* lds_ptr, const void* gptr) {
  unsigned loff = (unsigned)(unsigned long long)lds_ptr;
  asm volatile("global_load_async_to_lds_b128 %0, %1, off"
               :: "v"(loff), "v"(gptr)
               : "memory");
}
static __device__ __forceinline__ void wait_async0() {
  asm volatile("s_wait_asynccnt 0" ::: "memory");
}

// A-matrix fragment (16x32 bf16): lanes 0-15 -> M=lane, K = {koff..+7, 16+koff..+7}
static __device__ __forceinline__ bf16x16 frag_a(const __bf16* rowptr, int koff8) {
  union { bf16x16 v; bf16x8 h[2]; } u;
  u.h[0] = *(const bf16x8*)(rowptr + koff8);
  u.h[1] = *(const bf16x8*)(rowptr + 16 + koff8);
  return u.v;
}

// B-matrix fragment (32x16 bf16), weights pre-transposed to [N][K]:
// lanes 0-15 -> N=lane holds K={0..7,16..23}; lanes 16-31 -> K={8..15,24..31}
static __device__ __forceinline__ bf16x16 frag_b(const __bf16* rowptr, int kbase16) {
  union { bf16x16 v; bf16x8 h[2]; } u;
  u.h[0] = *(const bf16x8*)(rowptr + (kbase16 >> 1));
  u.h[1] = *(const bf16x8*)(rowptr + 16 + (kbase16 >> 1));
  return u.v;
}

// ---------- prep kernels ----------
__global__ void cvt_bf16(const float* __restrict__ src, __bf16* __restrict__ dst, int n) {
  int base = (blockIdx.x * blockDim.x + threadIdx.x) * 4;
  if (base + 3 < n) {
    float4 f = *(const float4*)(src + base);
    dst[base + 0] = to_bf16(f.x);
    dst[base + 1] = to_bf16(f.y);
    dst[base + 2] = to_bf16(f.z);
    dst[base + 3] = to_bf16(f.w);
  }
}

// dst[n][k] = bf16(src[k][n]) ; src is [K][N] row-major
__global__ void tconv_bf16(const float* __restrict__ src, __bf16* __restrict__ dst,
                           int K, int N) {
  int i = blockIdx.x * blockDim.x + threadIdx.x;
  if (i >= K * N) return;
  int n = i / K, k = i - n * K;
  dst[i] = to_bf16(src[(size_t)k * N + n]);
}

// ---------- fused recurrent GEMM + GRU gating ----------
__global__ __launch_bounds__(256) void gemm_gate(
    const __bf16* __restrict__ Ah,          // [NB][896] bf16 prev_hidden
    const __bf16* __restrict__ WtR,         // [2688][896] bf16 (W_R^T)
    const float*  __restrict__ prev_y,      // [NB][2]
    const float*  __restrict__ prev_hidden, // [NB][896] f32
    const float*  __restrict__ cur_c,       // [NB]
    const float*  __restrict__ W_Ic,        // [2][1344]
    const float*  __restrict__ W_If,        // [3][1344]
    const float*  __restrict__ bu,
    const float*  __restrict__ brr,
    const float*  __restrict__ bee,
    float*        __restrict__ hid_out,     // [NB][896] f32
    __bf16*       __restrict__ Hh)          // [NB][896] bf16
{
  __shared__ __align__(16) __bf16 As[2][128][40];
  __shared__ __align__(16) __bf16 Bs[2][3][64][40];

  const int tid  = threadIdx.x;
  const int lane = tid & 31;
  const int w    = tid >> 5;
  const int wm   = w & 3;        // 4 waves over M
  const int wn   = w >> 2;       // 2 waves over N
  const int m0   = blockIdx.x * 128;
  const int n0   = blockIdx.y * 64;
  const int lm   = lane & 15;
  const int hi16 = lane >> 4;

  // per-thread fill coordinates (16B chunks)
  const int ra = tid >> 2,          ca = (tid & 3) * 8;          // A chunk 1
  const int rా = (tid + 256) >> 2;                                // (unused marker)
  const int ra2 = (tid + 256) >> 2, ca2 = (tid & 3) * 8;          // A chunk 2
  const int rb = tid >> 2,          cb = (tid & 3) * 8;           // B chunk

  auto issue_tiles = [&](int buf, int k0) {
    async_ld_b128(&As[buf][ra][ca],   Ah + (size_t)(m0 + ra) * 896 + k0 + ca);
    async_ld_b128(&As[buf][ra2][ca2], Ah + (size_t)(m0 + ra2) * 896 + k0 + ca2);
#pragma unroll
    for (int g = 0; g < 3; ++g)
      async_ld_b128(&Bs[buf][g][rb][cb],
                    WtR + (size_t)(g * 896 + n0 + rb) * 896 + k0 + cb);
  };

  f32x8 acc[3][2][2] = {};

  issue_tiles(0, 0);
  int buf = 0;
  for (int k0 = 0; k0 < 896; k0 += 32) {
    wait_async0();
    __syncthreads();
    if (k0 + 32 < 896) issue_tiles(buf ^ 1, k0 + 32);

    bf16x16 aF[2];
#pragma unroll
    for (int mi = 0; mi < 2; ++mi)
      aF[mi] = frag_a(&As[buf][wm * 32 + mi * 16 + lm][0], hi16 * 8);
#pragma unroll
    for (int g = 0; g < 3; ++g) {
      bf16x16 bF[2];
#pragma unroll
      for (int ni = 0; ni < 2; ++ni)
        bF[ni] = frag_b(&Bs[buf][g][wn * 32 + ni * 16 + lm][0], hi16 * 16);
#pragma unroll
      for (int mi = 0; mi < 2; ++mi)
#pragma unroll
        for (int ni = 0; ni < 2; ++ni)
          acc[g][mi][ni] = wmma_bf16(aF[mi], bF[ni], acc[g][mi][ni]);
    }
    buf ^= 1;
  }

  // ---- epilogue: per-lane column constants ----
  int   hcol[2];
  float wu[2][3], wr[2][3], we[2][3], bU[2], bR[2], bE[2];
#pragma unroll
  for (int ni = 0; ni < 2; ++ni) {
    int h = n0 + wn * 32 + ni * 16 + lm;
    hcol[ni] = h;
    bU[ni] = bu[h]; bR[ni] = brr[h]; bE[ni] = bee[h];
    bool coarse = h < NSP;
    int c = coarse ? h : h - NSP;
    const float* Wp = coarse ? W_Ic : W_If;
    wu[ni][0] = Wp[c];        wu[ni][1] = Wp[1344 + c];
    wr[ni][0] = Wp[448 + c];  wr[ni][1] = Wp[1344 + 448 + c];
    we[ni][0] = Wp[896 + c];  we[ni][1] = Wp[1344 + 896 + c];
    wu[ni][2] = coarse ? 0.f : W_If[2688 + c];
    wr[ni][2] = coarse ? 0.f : W_If[2688 + 448 + c];
    we[ni][2] = coarse ? 0.f : W_If[2688 + 896 + c];
  }

#pragma unroll
  for (int mi = 0; mi < 2; ++mi) {
#pragma unroll
    for (int v = 0; v < 8; ++v) {
      int m = m0 + wm * 32 + mi * 16 + hi16 * 8 + v;
      float py0 = prev_y[2 * m], py1 = prev_y[2 * m + 1], cc = cur_c[m];
#pragma unroll
      for (int ni = 0; ni < 2; ++ni) {
        float Ru = acc[0][mi][ni][v];
        float Rr = acc[1][mi][ni][v];
        float Re = acc[2][mi][ni][v];
        float iu = fmaf(cc, wu[ni][2], fmaf(py1, wu[ni][1], py0 * wu[ni][0]));
        float ir = fmaf(cc, wr[ni][2], fmaf(py1, wr[ni][1], py0 * wr[ni][0]));
        float ie = fmaf(cc, we[ni][2], fmaf(py1, we[ni][1], py0 * we[ni][0]));
        float ug = 1.f / (1.f + __expf(-(Ru + iu + bU[ni])));
        float rg = 1.f / (1.f + __expf(-(Rr + ir + bR[ni])));
        float eg = tanhf(fmaf(rg, Re, ie + bE[ni]));
        float ph = prev_hidden[(size_t)m * 896 + hcol[ni]];
        float hd = fmaf(ug, ph - eg, eg);   // u*ph + (1-u)*e
        hid_out[(size_t)m * 896 + hcol[ni]] = hd;
        Hh[(size_t)m * 896 + hcol[ni]]      = to_bf16(hd);
      }
    }
  }
}

// ---------- generic head GEMM (128x64 block, 8 waves) ----------
// EPI==0: out_bf16 = bf16(relu(acc + bias))   EPI==1: out_f32 = acc + bias
template <int EPI>
__global__ __launch_bounds__(256) void gemm_head(
    const __bf16* __restrict__ A, int lda,
    const __bf16* __restrict__ Wt, int K,     // Wt is [N][K]
    const float*  __restrict__ bias,
    __bf16*       __restrict__ outb,
    float*        __restrict__ outf, int ldo)
{
  __shared__ __align__(16) __bf16 As[2][128][40];
  __shared__ __align__(16) __bf16 Bs[2][64][40];

  const int tid  = threadIdx.x;
  const int lane = tid & 31;
  const int w    = tid >> 5;
  const int wm   = w & 3;
  const int wn   = w >> 2;
  const int m0   = blockIdx.x * 128;
  const int n0   = blockIdx.y * 64;
  const int lm   = lane & 15;
  const int hi16 = lane >> 4;

  const int ra  = tid >> 2,         ca = (tid & 3) * 8;
  const int ra2 = (tid + 256) >> 2;
  const int rb  = tid >> 2,         cb = (tid & 3) * 8;

  auto issue_tiles = [&](int buf, int k0) {
    async_ld_b128(&As[buf][ra][ca],  A  + (size_t)(m0 + ra)  * lda + k0 + ca);
    async_ld_b128(&As[buf][ra2][ca], A  + (size_t)(m0 + ra2) * lda + k0 + ca);
    async_ld_b128(&Bs[buf][rb][cb],  Wt + (size_t)(n0 + rb)  * K   + k0 + cb);
  };

  f32x8 acc[2][2] = {};

  issue_tiles(0, 0);
  int buf = 0;
  for (int k0 = 0; k0 < K; k0 += 32) {
    wait_async0();
    __syncthreads();
    if (k0 + 32 < K) issue_tiles(buf ^ 1, k0 + 32);

    bf16x16 aF[2], bF[2];
#pragma unroll
    for (int mi = 0; mi < 2; ++mi)
      aF[mi] = frag_a(&As[buf][wm * 32 + mi * 16 + lm][0], hi16 * 8);
#pragma unroll
    for (int ni = 0; ni < 2; ++ni)
      bF[ni] = frag_b(&Bs[buf][wn * 32 + ni * 16 + lm][0], hi16 * 16);
#pragma unroll
    for (int mi = 0; mi < 2; ++mi)
#pragma unroll
      for (int ni = 0; ni < 2; ++ni)
        acc[mi][ni] = wmma_bf16(aF[mi], bF[ni], acc[mi][ni]);
    buf ^= 1;
  }

#pragma unroll
  for (int mi = 0; mi < 2; ++mi) {
#pragma unroll
    for (int ni = 0; ni < 2; ++ni) {
      int n  = n0 + wn * 32 + ni * 16 + lm;
      int mB = m0 + wm * 32 + mi * 16 + hi16 * 8;
      float bv = bias[n];
#pragma unroll
      for (int v = 0; v < 8; ++v) {
        float x = acc[mi][ni][v] + bv;
        if (EPI == 0) {
          x = fmaxf(x, 0.f);
          outb[(size_t)(mB + v) * ldo + n] = to_bf16(x);
        } else {
          outf[(size_t)(mB + v) * ldo + n] = x;
        }
      }
    }
  }
}

// ---------- launch ----------
extern "C" void kernel_launch(void* const* d_in, const int* in_sizes, int n_in,
                              void* d_out, int out_size, void* d_ws, size_t ws_size,
                              hipStream_t stream) {
  const float* prev_y      = (const float*)d_in[0];
  const float* prev_hidden = (const float*)d_in[1];
  const float* cur_c       = (const float*)d_in[2];
  const float* W_R         = (const float*)d_in[3];
  const float* W_Ic        = (const float*)d_in[4];
  const float* W_If        = (const float*)d_in[5];
  const float* bu          = (const float*)d_in[6];
  const float* br          = (const float*)d_in[7];
  const float* be          = (const float*)d_in[8];
  const float* W1          = (const float*)d_in[9];
  const float* b1          = (const float*)d_in[10];
  const float* W2          = (const float*)d_in[11];
  const float* b2          = (const float*)d_in[12];
  const float* W3          = (const float*)d_in[13];
  const float* b3          = (const float*)d_in[14];
  const float* W4          = (const float*)d_in[15];
  const float* b4          = (const float*)d_in[16];

  float* out = (float*)d_out;
  float* oc  = out;                            // out_coarse [NB][256]
  float* of  = out + (size_t)NB * NQ;          // out_fine   [NB][256]
  float* oh  = out + (size_t)NB * NQ * 2;      // hidden     [NB][896]

  char* ws = (char*)d_ws;
  __bf16* Ah  = (__bf16*)(ws);                            // 58,720,256 B
  __bf16* Hh  = (__bf16*)(ws + 58720256);                 // 58,720,256 B
  __bf16* T   = (__bf16*)(ws + 117440512);                // 29,360,128 B
  __bf16* WtR = (__bf16*)(ws + 146800640);                //  4,816,896 B
  __bf16* Wt1 = (__bf16*)(ws + 151617536);                //    401,408 B
  __bf16* Wt2 = (__bf16*)(ws + 152018944);                //    229,376 B
  __bf16* Wt3 = (__bf16*)(ws + 152248320);                //    401,408 B
  __bf16* Wt4 = (__bf16*)(ws + 152649728);                //    229,376 B

  cvt_bf16<<<(NB * NH / 4 + 255) / 256, 256, 0, stream>>>(prev_hidden, Ah, NB * NH);
  tconv_bf16<<<(2688 * 896 + 255) / 256, 256, 0, stream>>>(W_R, WtR, 896, 2688);
  tconv_bf16<<<(448 * 448 + 255) / 256, 256, 0, stream>>>(W1, Wt1, 448, 448);
  tconv_bf16<<<(448 * 256 + 255) / 256, 256, 0, stream>>>(W2, Wt2, 448, 256);
  tconv_bf16<<<(448 * 448 + 255) / 256, 256, 0, stream>>>(W3, Wt3, 448, 448);
  tconv_bf16<<<(448 * 256 + 255) / 256, 256, 0, stream>>>(W4, Wt4, 448, 256);

  gemm_gate<<<dim3(NB / 128, NH / 64), 256, 0, stream>>>(
      Ah, WtR, prev_y, prev_hidden, cur_c, W_Ic, W_If, bu, br, be, oh, Hh);

  gemm_head<0><<<dim3(NB / 128, NSP / 64), 256, 0, stream>>>(
      Hh, NH, Wt1, NSP, b1, T, nullptr, NSP);
  gemm_head<1><<<dim3(NB / 128, NQ / 64), 256, 0, stream>>>(
      T, NSP, Wt2, NSP, b2, nullptr, oc, NQ);

  gemm_head<0><<<dim3(NB / 128, NSP / 64), 256, 0, stream>>>(
      Hh + NSP, NH, Wt3, NSP, b3, T, nullptr, NSP);
  gemm_head<1><<<dim3(NB / 128, NQ / 64), 256, 0, stream>>>(
      T, NSP, Wt4, NSP, b4, nullptr, of, NQ);
}
